// HMM_76682346103353
// MI455X (gfx1250) — matrix operand order repaired and checked by reading, not compiled, not run on hardware
//
#include <hip/hip_runtime.h>
#include <hip/hip_bf16.h>
#include <math.h>

#define H 2048
#define V 32000
#define B 16
#define S 256

typedef __attribute__((ext_vector_type(8)))  __bf16 v8bf;
typedef __attribute__((ext_vector_type(16))) __bf16 v16bf;
typedef __attribute__((ext_vector_type(8)))  float  v8f;

// ---------------------------------------------------------------------------
// 1) Gather emissions: ys[k][h][b] = beta[h, input_ids[b, S-1-k]]
//    (ys[0] is already the final y0; later steps update ys[k] in place.)
// ---------------------------------------------------------------------------
__global__ void k_gather(const int* __restrict__ ids,
                         const float* __restrict__ beta,
                         float* __restrict__ ys) {
    size_t i = (size_t)blockIdx.x * blockDim.x + threadIdx.x;  // < S*H*B
    int b = (int)(i % B);
    size_t hk = i / B;
    int h = (int)(hk % H);
    int k = (int)(hk / H);
    int t = S - 1 - k;
    int id = ids[b * S + t];
    ys[i] = beta[(size_t)h * V + id];
}

// ---------------------------------------------------------------------------
// 2) Convert alpha_exp (f32 row-major HxH) to bf16 once.
// ---------------------------------------------------------------------------
__global__ void k_cvt(const float* __restrict__ a, __bf16* __restrict__ ab) {
    size_t i = (size_t)blockIdx.x * blockDim.x + threadIdx.x;
    ab[i] = (__bf16)a[i];
}

// ---------------------------------------------------------------------------
// 3) Per-step: column-wise max over H, then eB[b*H+h] = bf16(exp(y-mx))
//    (column-major so WMMA B-fragments are contiguous loads). One WG.
// ---------------------------------------------------------------------------
__global__ void __launch_bounds__(1024)
k_rexp(const float* __restrict__ yprev, float* __restrict__ ymax,
       __bf16* __restrict__ eB) {
    __shared__ float red[1024];
    int tid = threadIdx.x;
    int b = tid & 15;
    int seg = tid >> 4;  // 0..63
    float m = -INFINITY;
    #pragma unroll 4
    for (int j = 0; j < H / 64; ++j) {
        int h = seg + j * 64;                     // coalesced: addr = tid + 1024*j
        m = fmaxf(m, yprev[h * B + b]);
    }
    red[tid] = m;
    __syncthreads();
    for (int s = 512; s >= 16; s >>= 1) {
        if (tid < s) red[tid] = fmaxf(red[tid], red[tid + s]);
        __syncthreads();
    }
    if (tid < 16) ymax[tid] = red[tid];
    // phase 2: write exp() column-major; each thread: 32 contiguous elements
    int b2 = tid >> 6;            // 0..15
    int hs = (tid & 63) * 32;     // row start
    float mb = red[b2];
    #pragma unroll 4
    for (int j = 0; j < 32; ++j) {
        int h = hs + j;
        eB[(size_t)b2 * H + h] = (__bf16)__expf(yprev[h * B + b2] - mb);
    }
}

// ---------------------------------------------------------------------------
// 4) Per-step GEMM: D(2048x16) = alpha_bf16(2048x2048) @ eB(2048x16), then
//    ys[k] = log(D) + ymax + ys[k]   (ys[k] currently holds the emission ip).
//    128 WGs x 128 threads; 4 waves split-K (512 each), LDS combine.
// ---------------------------------------------------------------------------
__global__ void __launch_bounds__(128)
k_gemm(const __bf16* __restrict__ alpha, const __bf16* __restrict__ eB,
       const float* __restrict__ ymax, float* __restrict__ yk) {
    __shared__ float lds_acc[4][256];
    int wave = threadIdx.x >> 5;
    int lane = threadIdx.x & 31;
    int half = lane >> 4;       // 0: lanes 0-15, 1: lanes 16-31
    int ln   = lane & 15;       // row (A) / column (B,D) index
    int mtile = blockIdx.x * 16;

    const __bf16* arow = alpha + (size_t)(mtile + ln) * H;  // A row per lane
    const __bf16* bcol = eB + (size_t)ln * H;               // B column per lane
    const int aoff = half ? 8 : 0;    // A K-offsets: {0..7,16..23} / {8..15,24..31}
    const int boff = half ? 8 : 0;    // B K-offsets: {0..7,16..23} / {8..15,24..31}

    v8f acc = {};
    int k0 = wave * (H / 4);
    #pragma unroll 4
    for (int kb = k0; kb < k0 + H / 4; kb += 32) {
        v8bf alo = *(const v8bf*)(arow + kb + aoff);
        v8bf ahi = *(const v8bf*)(arow + kb + aoff + 16);
        v8bf blo = *(const v8bf*)(bcol + kb + boff);
        v8bf bhi = *(const v8bf*)(bcol + kb + boff + 16);
        v16bf Af = __builtin_shufflevector(alo, ahi,
                      0,1,2,3,4,5,6,7,8,9,10,11,12,13,14,15);
        v16bf Bf = __builtin_shufflevector(blo, bhi,
                      0,1,2,3,4,5,6,7,8,9,10,11,12,13,14,15);
        acc = __builtin_amdgcn_wmma_f32_16x16x32_bf16(
                  false, Af, false, Bf, (short)0, acc, false, false);
    }
    #pragma unroll
    for (int r = 0; r < 8; ++r) lds_acc[wave][r * 32 + lane] = acc[r];
    __syncthreads();

    if (wave == 0) {
        float mb = ymax[ln];
        #pragma unroll
        for (int r = 0; r < 8; ++r) {
            int e = r * 32 + lane;
            float s = lds_acc[0][e] + lds_acc[1][e] + lds_acc[2][e] + lds_acc[3][e];
            int m = mtile + r + half * 8;                 // D layout: M = r (+8)
            size_t idx = (size_t)m * B + ln;              // N = ln
            yk[idx] = __logf(s) + mb + yk[idx];
        }
    }
}

// ---------------------------------------------------------------------------
// 5) Final: final[b] = log(sum_h softmax(gamma)[h] * exp(y-mx)) + mx
//        = log(sum_h exp(gamma[h]-gmax + y[h,b]-mx_b)) - log(gsum) + mx_b
// ---------------------------------------------------------------------------
__global__ void __launch_bounds__(1024)
k_final(const float* __restrict__ ylast, const float* __restrict__ gamma,
        float* __restrict__ out) {
    __shared__ float red[1024];
    int tid = threadIdx.x;
    // gamma max
    float g0 = gamma[tid], g1 = gamma[tid + 1024];
    red[tid] = fmaxf(g0, g1);
    __syncthreads();
    for (int s = 512; s >= 1; s >>= 1) {
        if (tid < s) red[tid] = fmaxf(red[tid], red[tid + s]);
        __syncthreads();
    }
    float gmax = red[0];
    __syncthreads();
    red[tid] = __expf(g0 - gmax) + __expf(g1 - gmax);
    __syncthreads();
    for (int s = 512; s >= 1; s >>= 1) {
        if (tid < s) red[tid] += red[tid + s];
        __syncthreads();
    }
    float gsum = red[0];
    __syncthreads();
    // per-column ymax of y_last
    int b = tid & 15, seg = tid >> 4;
    float m = -INFINITY;
    for (int j = 0; j < H / 64; ++j) {
        int h = seg + j * 64;
        m = fmaxf(m, ylast[h * B + b]);
    }
    red[tid] = m;
    __syncthreads();
    for (int s = 512; s >= 16; s >>= 1) {
        if (tid < s) red[tid] = fmaxf(red[tid], red[tid + s]);
        __syncthreads();
    }
    float mb = red[b];
    __syncthreads();
    float ssum = 0.f;
    for (int j = 0; j < H / 64; ++j) {
        int h = seg + j * 64;
        ssum += __expf(gamma[h] - gmax + ylast[h * B + b] - mb);
    }
    red[tid] = ssum;
    __syncthreads();
    for (int s = 512; s >= 16; s >>= 1) {
        if (tid < s) red[tid] += red[tid + s];
        __syncthreads();
    }
    if (tid < 16) out[tid] = __logf(red[tid]) - __logf(gsum) + mb;
}

// ---------------------------------------------------------------------------
extern "C" void kernel_launch(void* const* d_in, const int* in_sizes, int n_in,
                              void* d_out, int out_size, void* d_ws, size_t ws_size,
                              hipStream_t stream) {
    const int*   ids   = (const int*)d_in[0];    // (B,S)
    const float* alpha = (const float*)d_in[1];  // (H,H)
    const float* beta  = (const float*)d_in[2];  // (H,V)
    const float* gamma = (const float*)d_in[3];  // (H,)

    float* ys = (float*)d_out;                        // (S,H,B)
    float* final_out = ys + (size_t)S * H * B;        // (B,)

    char* ws = (char*)d_ws;
    __bf16* alpha_bf = (__bf16*)ws;                               // 8 MB
    float*  ymax     = (float*)(ws + (size_t)H * H * 2);          // 64 B
    __bf16* eB       = (__bf16*)(ws + (size_t)H * H * 2 + 256);   // 64 KB

    k_cvt<<<(H * H) / 256, 256, 0, stream>>>(alpha, alpha_bf);
    k_gather<<<((size_t)S * H * B) / 256, 256, 0, stream>>>(ids, beta, ys);

    for (int k = 1; k < S; ++k) {
        const float* yprev = ys + (size_t)(k - 1) * H * B;
        float*       yk    = ys + (size_t)k * H * B;
        k_rexp<<<1, 1024, 0, stream>>>(yprev, ymax, eB);
        k_gemm<<<H / 16, 128, 0, stream>>>(alpha_bf, eB, ymax, yk);
    }

    k_final<<<1, 1024, 0, stream>>>(ys + (size_t)(S - 1) * H * B, gamma, final_out);
}
